// Attention_6983616824195
// MI455X (gfx1250) — matrix operand order repaired and checked by reading, not compiled, not run on hardware
//
#include <hip/hip_runtime.h>
#include <hip/hip_bf16.h>
#include <math.h>

typedef __attribute__((ext_vector_type(16))) __bf16 v16bf;
typedef __attribute__((ext_vector_type(8)))  __bf16 v8bf;
typedef __attribute__((ext_vector_type(4)))  __bf16 v4bf;
typedef __attribute__((ext_vector_type(8)))  float  v8f;
typedef __attribute__((ext_vector_type(4)))  float  v4f;

#define BATCH  4
#define SEQ    4096
#define DMODEL 1024
#define DHEAD  128
#define FA_WAVES 4

__device__ __forceinline__ v16bf combine(v8bf lo, v8bf hi) {
  v16bf r;
#pragma unroll
  for (int i = 0; i < 8; i++) { r[i] = lo[i]; r[i + 8] = hi[i]; }
  return r;
}

// ---------------------------------------------------------------------------
// Weight prep: W[1024,128] f32 -> Wt[128,1024] bf16 (B-frag friendly: each
// output column's K-run is contiguous).  Tiny one-shot pass; W lives in L2.
// ---------------------------------------------------------------------------
__global__ void __launch_bounds__(256)
transpose_w(const float* __restrict__ W, __bf16* __restrict__ Wt) {
  int idx = blockIdx.x * 256 + threadIdx.x;      // 0 .. 1024*128-1
  int kk = idx >> 7;
  int c  = idx & 127;
  Wt[(size_t)c * DMODEL + kk] = (__bf16)W[idx];
}

// ---------------------------------------------------------------------------
// Projection: Y[BL,128](bf16) = X[BL,1024](f32) @ W (via Wt bf16 [128,1024]).
// Block = 16 rows x 128 cols, 8 waves (wave w owns column tile w).
// X tile staged once in LDS as bf16 (shared A-frags, single f32->bf16 pass).
// transpose_out=1 stores Y^T per batch: vt[b][col][s] (for the PV stage).
// ---------------------------------------------------------------------------
__global__ void __launch_bounds__(256)
proj_kernel(const float* __restrict__ X, const __bf16* __restrict__ Wt,
            __bf16* __restrict__ Y, int transpose_out) {
  __shared__ __bf16 xtile[16][DMODEL];           // 32 KB

  const int tid  = threadIdx.x;
  const int lane = tid & 31;
  const int wave = tid >> 5;                     // 0..7 -> column tile
  const int r0   = blockIdx.x * 16;              // global row base
  const int c0   = wave * 16;
  const int m    = lane & 15;
  const int half = lane >> 4;

  // Cooperative stage: 16x1024 f32 -> bf16 LDS (each thread: 16 x float4).
#pragma unroll
  for (int i = tid * 4; i < 16 * DMODEL; i += 256 * 4) {
    int row = i >> 10, col = i & (DMODEL - 1);
    v4f x = *(const v4f*)(X + (size_t)(r0 + row) * DMODEL + col);
    v4bf xb;
#pragma unroll
    for (int j = 0; j < 4; j++) xb[j] = (__bf16)x[j];
    *(v4bf*)&xtile[row][col] = xb;
  }
  __syncthreads();

  v8f acc = {};
#pragma unroll 4
  for (int kc = 0; kc < DMODEL; kc += 32) {
    // A fragment (16x32) from LDS: two 16-byte ds loads.
    const __bf16* arow = &xtile[m][kc + half * 16];
    v16bf afrag = combine(*(const v8bf*)arow, *(const v8bf*)(arow + 8));
    // B fragment (32x16) from Wt: contiguous K-run, two global b128 loads.
    const __bf16* wrow = Wt + (size_t)(c0 + m) * DMODEL + kc + half * 16;
    v16bf bfrag = combine(*(const v8bf*)wrow, *(const v8bf*)(wrow + 8));

    acc = __builtin_amdgcn_wmma_f32_16x16x32_bf16(
        false, afrag, false, bfrag, (short)0, acc, false, false);
  }

  if (!transpose_out) {
#pragma unroll
    for (int i = 0; i < 8; i++) {
      int row = i + half * 8;
      Y[(size_t)(r0 + row) * DHEAD + c0 + m] = (__bf16)acc[i];
    }
  } else {
#pragma unroll
    for (int i = 0; i < 8; i++) {
      int grow = r0 + i + half * 8;
      int b = grow >> 12;                        // / SEQ
      int s = grow & (SEQ - 1);
      Y[(size_t)b * DHEAD * SEQ + (size_t)(c0 + m) * SEQ + s] = (__bf16)acc[i];
    }
  }
}

// ---------------------------------------------------------------------------
// Flash attention: each wave owns 16 query rows, streams keys in chunks of 32.
// q,k: bf16 [B*L,128] row-major.  vt: bf16 [B][128][L] (transposed).
// ---------------------------------------------------------------------------
__global__ void __launch_bounds__(FA_WAVES * 32)
flash_kernel(const __bf16* __restrict__ q, const __bf16* __restrict__ k,
             const __bf16* __restrict__ vt, const int* __restrict__ mask,
             float* __restrict__ out) {
  __shared__ __bf16 plds[FA_WAVES][16][32];      // per-wave P transpose scratch

  const int lane = threadIdx.x & 31;
  const int wave = threadIdx.x >> 5;
  const int m    = lane & 15;
  const int half = lane >> 4;
  const int grow0 = blockIdx.x * (FA_WAVES * 16) + wave * 16; // global q row
  const int b     = grow0 >> 12;
  const int srow0 = b << 12;                                  // batch row base
  const float NEGINF = -__builtin_inff();

  // Preload Q A-fragments for all 4 K-chunks (kept in registers all kernel).
  v16bf qf[4];
#pragma unroll
  for (int kc = 0; kc < 4; kc++) {
    const __bf16* arow = q + (size_t)(grow0 + m) * DHEAD + kc * 32 + half * 16;
    qf[kc] = combine(*(const v8bf*)arow, *(const v8bf*)(arow + 8));
  }

  v8f o[8];
#pragma unroll
  for (int i = 0; i < 8; i++) { v8f z = {}; o[i] = z; }
  float mrow[8], lrow[8];
#pragma unroll
  for (int i = 0; i < 8; i++) { mrow[i] = NEGINF; lrow[i] = 0.0f; }

  const float scale = 0.08838834764831845f;      // 1/sqrt(128)
  const int* mptr = mask + (size_t)b * SEQ;

  for (int s0 = 0; s0 < SEQ; s0 += 32) {
    // ---- S = q @ k^T over 32 keys (two 16x16 C tiles) ----
    v8f sA = {}, sB = {};
    const __bf16* k0 = k + (size_t)(srow0 + s0 + m) * DHEAD;  // B col = key row
    const __bf16* k1 = k0 + (size_t)16 * DHEAD;
    __builtin_prefetch(k0 + (size_t)32 * DHEAD, 0, 1);        // next key chunk
#pragma unroll
    for (int kc = 0; kc < 4; kc++) {
      int koff = kc * 32 + half * 16;
      v16bf bA = combine(*(const v8bf*)(k0 + koff), *(const v8bf*)(k0 + koff + 8));
      sA = __builtin_amdgcn_wmma_f32_16x16x32_bf16(false, qf[kc], false, bA,
                                                   (short)0, sA, false, false);
      v16bf bB = combine(*(const v8bf*)(k1 + koff), *(const v8bf*)(k1 + koff + 8));
      sB = __builtin_amdgcn_wmma_f32_16x16x32_bf16(false, qf[kc], false, bB,
                                                   (short)0, sB, false, false);
    }

    // ---- mask (before scaling, as in reference) then scale ----
    const int mk0 = mptr[s0 + m];
    const int mk1 = mptr[s0 + 16 + m];
#pragma unroll
    for (int i = 0; i < 8; i++) {
      sA[i] = (mk0 == 1) ? sA[i] * scale : NEGINF;
      sB[i] = (mk1 == 1) ? sB[i] * scale : NEGINF;
    }

    // ---- online softmax (row stats replicated across 16-lane halves) ----
    float alpha[8];
#pragma unroll
    for (int i = 0; i < 8; i++) {
      float rmax = fmaxf(sA[i], sB[i]);
#pragma unroll
      for (int d = 1; d < 16; d <<= 1) rmax = fmaxf(rmax, __shfl_xor(rmax, d, 32));
      float mnew = fmaxf(mrow[i], rmax);
      float a, pA, pB;
      if (mnew == NEGINF) { a = 1.0f; pA = 0.0f; pB = 0.0f; }
      else {
        a  = __expf(mrow[i] - mnew);
        pA = __expf(sA[i] - mnew);
        pB = __expf(sB[i] - mnew);
      }
      float rsum = pA + pB;
#pragma unroll
      for (int d = 1; d < 16; d <<= 1) rsum += __shfl_xor(rsum, d, 32);
      lrow[i]  = lrow[i] * a + rsum;
      mrow[i]  = mnew;
      alpha[i] = a;
      sA[i] = pA; sB[i] = pB;
    }
#pragma unroll
    for (int nt = 0; nt < 8; nt++)
#pragma unroll
      for (int i = 0; i < 8; i++) o[nt][i] *= alpha[i];

    // ---- P: C-layout -> A-layout via per-wave LDS scratch ----
#pragma unroll
    for (int i = 0; i < 8; i++) {
      int row = i + half * 8;
      plds[wave][row][m]      = (__bf16)sA[i];
      plds[wave][row][16 + m] = (__bf16)sB[i];
    }
    const __bf16* prow = &plds[wave][m][half * 16];
    v16bf pf = combine(*(const v8bf*)prow, *(const v8bf*)(prow + 8));

    // ---- O += P @ V (vt transposed: B-frag K-run contiguous) ----
    const __bf16* vcol = vt + (size_t)b * DHEAD * SEQ + (size_t)m * SEQ
                         + s0 + half * 16;
#pragma unroll
    for (int nt = 0; nt < 8; nt++) {
      const __bf16* vp = vcol + (size_t)(nt * 16) * SEQ;
      v16bf vf = combine(*(const v8bf*)vp, *(const v8bf*)(vp + 8));
      o[nt] = __builtin_amdgcn_wmma_f32_16x16x32_bf16(false, pf, false, vf,
                                                      (short)0, o[nt], false, false);
    }
  }

  // ---- normalize and store f32 output ----
#pragma unroll
  for (int i = 0; i < 8; i++) {
    float inv = (lrow[i] > 0.0f) ? 1.0f / lrow[i] : 0.0f;
    int row = grow0 + i + half * 8;
#pragma unroll
    for (int nt = 0; nt < 8; nt++)
      out[(size_t)row * DHEAD + nt * 16 + m] = o[nt][i] * inv;
  }
}

// ---------------------------------------------------------------------------
extern "C" void kernel_launch(void* const* d_in, const int* in_sizes, int n_in,
                              void* d_out, int out_size, void* d_ws, size_t ws_size,
                              hipStream_t stream) {
  (void)in_sizes; (void)n_in; (void)out_size; (void)ws_size;
  const float* Q    = (const float*)d_in[0];
  const float* K    = (const float*)d_in[1];
  const float* V    = (const float*)d_in[2];
  const int*   mask = (const int*)d_in[3];
  const float* WQ   = (const float*)d_in[4];
  const float* WK   = (const float*)d_in[5];
  const float* WV   = (const float*)d_in[6];
  float* out = (float*)d_out;

  const size_t QKV = (size_t)BATCH * SEQ * DHEAD;     // 2 MB elems -> 4 MB bf16
  const size_t WSZ = (size_t)DMODEL * DHEAD;          // 128K elems -> 256 KB bf16
  __bf16* qws = (__bf16*)d_ws;
  __bf16* kws = qws + QKV;
  __bf16* vws = kws + QKV;                            // stored transposed
  __bf16* wqt = vws + QKV;
  __bf16* wkt = wqt + WSZ;
  __bf16* wvt = wkt + WSZ;

  dim3 wb(256), wg(WSZ / 256);
  hipLaunchKernelGGL(transpose_w, wg, wb, 0, stream, WQ, wqt);
  hipLaunchKernelGGL(transpose_w, wg, wb, 0, stream, WK, wkt);
  hipLaunchKernelGGL(transpose_w, wg, wb, 0, stream, WV, wvt);

  dim3 pb(256), pg(BATCH * SEQ / 16);
  hipLaunchKernelGGL(proj_kernel, pg, pb, 0, stream, Q, wqt, qws, 0);
  hipLaunchKernelGGL(proj_kernel, pg, pb, 0, stream, K, wkt, kws, 0);
  hipLaunchKernelGGL(proj_kernel, pg, pb, 0, stream, V, wvt, vws, 1);

  dim3 fb(FA_WAVES * 32), fg(BATCH * SEQ / (FA_WAVES * 16));
  hipLaunchKernelGGL(flash_kernel, fg, fb, 0, stream, qws, kws, vws, mask, out);
}
